// DQRNN_10136122819197
// MI455X (gfx1250) — matrix operand order repaired and checked by reading, not compiled, not run on hardware
//
#include <hip/hip_runtime.h>
#include <hip/hip_bf16.h>
#include <stdint.h>
#include <stddef.h>

// ---------------------------------------------------------------------------
// DQRNN forward for MI455X (gfx1250): bf16 WMMA GEMMs + async-to-LDS pipeline
//  - fc1/fc2 + gi-precompute: 8-wave tiled GEMM, double-buffered LDS filled
//    with global_load_async_to_lds_b128 (ASYNCcnt / s_wait_asynccnt).
//  - GRU scan: 256 sequential step kernels; per block, h-tile staged once in
//    LDS and shared by 8 waves; only h@Whh^T remains on the critical path.
// ---------------------------------------------------------------------------

typedef __attribute__((ext_vector_type(16))) __bf16 v16bf;
typedef __attribute__((ext_vector_type(8)))  float  v8f;

#define T_DIM 256
#define B_DIM 256
#define I_DIM 256
#define H_DIM 512
#define G3H   (3 * H_DIM)
#define F1_DIM 1024
#define F2_DIM 1024
#define O_DIM 64

__device__ __forceinline__ unsigned short f32_to_bf16(float f) {
  unsigned u = __float_as_uint(f);
  u += 0x7FFFu + ((u >> 16) & 1u);           // round-to-nearest-even
  return (unsigned short)(u >> 16);
}
__device__ __forceinline__ float bf16_to_f32(unsigned short h) {
  return __uint_as_float(((unsigned)h) << 16);
}

union Frag {
  v16bf v;
  uint4 q[2];
};

// CDNA5 async copy: global -> LDS, tracked with ASYNCcnt (CDNA5 ISA)
__device__ __forceinline__ void async_b128(const void* g, const void* l) {
  unsigned     lds = (unsigned)(unsigned long long)l;   // flat LDS addr low 32b = WG-relative offset
  unsigned long long ga = (unsigned long long)g;
  asm volatile("global_load_async_to_lds_b128 %0, %1, off" :: "v"(lds), "v"(ga) : "memory");
}
template<int N>
__device__ __forceinline__ void wait_async() {
  asm volatile("s_wait_asynccnt %0" :: "i"(N) : "memory");
}

// ---------------------------------------------------------------------------
// Tiled GEMM: C[M,N](f32 accum) = A[M,K](bf16,row-major) * W[N,K](bf16,row-major)^T + bias
// ---------------------------------------------------------------------------
template<int BM, int BN>
__global__ __launch_bounds__(256) void gemm_bf16_kernel(
    const unsigned short* __restrict__ A, int lda,
    const unsigned short* __restrict__ Bw, int ldb,
    const float* __restrict__ bias,
    int M, int N, int K,
    unsigned short* __restrict__ Cbf, float* __restrict__ Cf,
    int relu)
{
  constexpr int BK  = 32;
  constexpr int WTM = BM / 2;          // wave grid 2 x 4
  constexpr int WTN = BN / 4;
  constexpr int MI  = WTM / 16;
  constexpr int NI  = WTN / 16;
  constexpr int CPF = (BM * 4) / 256 + (BN * 4) / 256;  // async ops per thread per stage

  __shared__ alignas(16) unsigned short sA[2][BM * BK];
  __shared__ alignas(16) unsigned short sB[2][BN * BK];

  const int tid  = threadIdx.x;
  const int wave = tid >> 5;
  const int lane = tid & 31;
  const int ln   = lane & 15;
  const int lh   = lane >> 4;
  const int wrow = wave >> 2;
  const int wcol = wave & 3;

  const int m0 = blockIdx.x * BM;
  const int n0 = blockIdx.y * BN;

  v8f acc[MI][NI];
#pragma unroll
  for (int mi = 0; mi < MI; ++mi)
#pragma unroll
    for (int ni = 0; ni < NI; ++ni)
      acc[mi][ni] = (v8f){0.f,0.f,0.f,0.f,0.f,0.f,0.f,0.f};

  auto fill = [&](int buf, int kb) {
#pragma unroll
    for (int i = 0; i < (BM * 4) / 256; ++i) {
      int idx = tid + i * 256;
      int r = idx >> 2, c = (idx & 3) * 8;
      async_b128(A + (size_t)(m0 + r) * lda + kb + c, &sA[buf][r * BK + c]);
    }
#pragma unroll
    for (int i = 0; i < (BN * 4) / 256; ++i) {
      int idx = tid + i * 256;
      int r = idx >> 2, c = (idx & 3) * 8;
      async_b128(Bw + (size_t)(n0 + r) * ldb + kb + c, &sB[buf][r * BK + c]);
    }
  };

  fill(0, 0);
  const int nk = K / BK;
  for (int kt = 0; kt < nk; ++kt) {
    const int buf = kt & 1;
    if (kt + 1 < nk) {
      fill(buf ^ 1, (kt + 1) * BK);   // prefetch next stage
      wait_async<CPF>();              // current stage complete (per-wave, in-order)
    } else {
      wait_async<0>();
    }
    __syncthreads();                  // make LDS stage visible to all 8 waves

    Frag fa[MI], fb[NI];
#pragma unroll
    for (int mi = 0; mi < MI; ++mi) {
      const unsigned short* pa = &sA[buf][(wrow * WTM + mi * 16 + ln) * BK + lh * 8];
      fa[mi].q[0] = *(const uint4*)(pa);        // K = kh*8 .. +7
      fa[mi].q[1] = *(const uint4*)(pa + 16);   // K = 16+kh*8 .. +7
    }
#pragma unroll
    for (int ni = 0; ni < NI; ++ni) {
      const unsigned short* pb = &sB[buf][(wcol * WTN + ni * 16 + ln) * BK + lh * 8];
      fb[ni].q[0] = *(const uint4*)(pb);
      fb[ni].q[1] = *(const uint4*)(pb + 16);
    }
#pragma unroll
    for (int mi = 0; mi < MI; ++mi)
#pragma unroll
      for (int ni = 0; ni < NI; ++ni)
        acc[mi][ni] = __builtin_amdgcn_wmma_f32_16x16x32_bf16(
            false, fa[mi].v, false, fb[ni].v, (short)0, acc[mi][ni], false, false);
    __syncthreads();                  // all reads done before this buffer is refilled
  }

#pragma unroll
  for (int mi = 0; mi < MI; ++mi) {
#pragma unroll
    for (int ni = 0; ni < NI; ++ni) {
      const int gn = n0 + wcol * WTN + ni * 16 + ln;
      const float bv = bias ? bias[gn] : 0.0f;
#pragma unroll
      for (int v = 0; v < 8; ++v) {
        const int gm = m0 + wrow * WTM + mi * 16 + v + 8 * lh;
        float f = acc[mi][ni][v] + bv;
        if (relu) f = fmaxf(f, 0.0f);
        if (Cbf) Cbf[(size_t)gm * N + gn] = f32_to_bf16(f);
        if (Cf)  Cf[(size_t)gm * N + gn]  = f;
      }
    }
  }
}

// ---------------------------------------------------------------------------
// GRU step (gi precomputed): 8 waves/block; block covers rows b0..b0+15 and
// 128 columns of H. The 16x512 h_prev tile is staged once into LDS (async)
// and shared by all 8 waves as the WMMA A operand. Per wave: 3 accumulators
// (r and z gates: gh only; n gate: gh kept separate for r*h_n).
// ---------------------------------------------------------------------------
__global__ __launch_bounds__(256) void gru_step_kernel(
    const unsigned short* __restrict__ git,    // [B,3H] bf16: gi for this timestep
    const unsigned short* __restrict__ hprev,  // [B,H] bf16
    const unsigned short* __restrict__ Whh,    // [3H,H] bf16 (L2-resident)
    const float* __restrict__ bih, const float* __restrict__ bhh,
    unsigned short* __restrict__ hnew)         // [B,H] bf16
{
  __shared__ alignas(16) unsigned short sH[16 * H_DIM];   // 16 rows x 512 cols = 16KB

  const int tid  = threadIdx.x;
  const int wave = tid >> 5;
  const int lane = tid & 31;
  const int ln   = lane & 15;
  const int lh   = lane >> 4;
  const int b0   = blockIdx.x * 16;
  const int j0   = blockIdx.y * 128 + wave * 16;

  // Stage h_prev tile: 1024 chunks of 8 bf16, 4 async b128 per thread.
#pragma unroll
  for (int i = 0; i < 4; ++i) {
    int idx = tid + i * 256;
    int r = idx >> 6;                 // 64 chunks per 512-elem row
    int c = (idx & 63) * 8;
    async_b128(hprev + (size_t)(b0 + r) * H_DIM + c, &sH[r * H_DIM + c]);
  }
  wait_async<0>();
  __syncthreads();

  v8f accR  = (v8f){0.f,0.f,0.f,0.f,0.f,0.f,0.f,0.f};
  v8f accZ  = accR, accHN = accR;

  const unsigned short* whhR = Whh + (size_t)(0 * H_DIM + j0 + ln) * H_DIM;
  const unsigned short* whhZ = Whh + (size_t)(1 * H_DIM + j0 + ln) * H_DIM;
  const unsigned short* whhN = Whh + (size_t)(2 * H_DIM + j0 + ln) * H_DIM;
  const unsigned short* arow = &sH[ln * H_DIM];

#pragma unroll 4
  for (int kb = 0; kb < H_DIM; kb += 32) {
    Frag ah, b;
    const unsigned short* pa = arow + kb + lh * 8;
    ah.q[0] = *(const uint4*)pa;  ah.q[1] = *(const uint4*)(pa + 16);

    const int ko = kb + lh * 8;
    b.q[0] = *(const uint4*)(whhR + ko); b.q[1] = *(const uint4*)(whhR + ko + 16);
    accR  = __builtin_amdgcn_wmma_f32_16x16x32_bf16(false, ah.v, false, b.v, (short)0, accR,  false, false);
    b.q[0] = *(const uint4*)(whhZ + ko); b.q[1] = *(const uint4*)(whhZ + ko + 16);
    accZ  = __builtin_amdgcn_wmma_f32_16x16x32_bf16(false, ah.v, false, b.v, (short)0, accZ,  false, false);
    b.q[0] = *(const uint4*)(whhN + ko); b.q[1] = *(const uint4*)(whhN + ko + 16);
    accHN = __builtin_amdgcn_wmma_f32_16x16x32_bf16(false, ah.v, false, b.v, (short)0, accHN, false, false);
  }

  const int j = j0 + ln;
  const float bihR = bih[j],             bhhR = bhh[j];
  const float bihZ = bih[H_DIM + j],     bhhZ = bhh[H_DIM + j];
  const float bihN = bih[2 * H_DIM + j], bhhN = bhh[2 * H_DIM + j];

#pragma unroll
  for (int v = 0; v < 8; ++v) {
    const int br = v + 8 * lh;                 // row within tile
    const int b  = b0 + br;
    const unsigned short* gib = git + (size_t)b * G3H;
    float giR = bf16_to_f32(gib[j]);
    float giZ = bf16_to_f32(gib[H_DIM + j]);
    float giN = bf16_to_f32(gib[2 * H_DIM + j]);
    float r = 1.0f / (1.0f + __expf(-(giR + bihR + accR[v] + bhhR)));
    float z = 1.0f / (1.0f + __expf(-(giZ + bihZ + accZ[v] + bhhZ)));
    float n = tanhf(giN + bihN + r * (accHN[v] + bhhN));
    float hp = bf16_to_f32(sH[br * H_DIM + j]);
    float hn = (1.0f - z) * n + z * hp;
    hnew[(size_t)b * H_DIM + j] = f32_to_bf16(hn);
  }
}

// ---------------------------------------------------------------------------
__global__ void f32_to_bf16_kernel(const float* __restrict__ s,
                                   unsigned short* __restrict__ d, size_t n) {
  size_t i = (size_t)blockIdx.x * blockDim.x + threadIdx.x;
  size_t st = (size_t)gridDim.x * blockDim.x;
  for (; i < n; i += st) d[i] = f32_to_bf16(s[i]);
}
__global__ void zero_bf16_kernel(unsigned short* __restrict__ d, size_t n) {
  size_t i = (size_t)blockIdx.x * blockDim.x + threadIdx.x;
  size_t st = (size_t)gridDim.x * blockDim.x;
  for (; i < n; i += st) d[i] = 0;
}

// ---------------------------------------------------------------------------
extern "C" void kernel_launch(void* const* d_in, const int* in_sizes, int n_in,
                              void* d_out, int out_size, void* d_ws, size_t ws_size,
                              hipStream_t stream) {
  const float* inputs = (const float*)d_in[0];
  const float* W1a = (const float*)d_in[1];
  const float* b1a = (const float*)d_in[2];
  const float* W1b = (const float*)d_in[3];
  const float* b1b = (const float*)d_in[4];
  const float* Wih = (const float*)d_in[5];
  const float* Whh = (const float*)d_in[6];
  const float* bih = (const float*)d_in[7];
  const float* bhh = (const float*)d_in[8];
  const float* W2a = (const float*)d_in[9];
  const float* b2a = (const float*)d_in[10];
  const float* W2b = (const float*)d_in[11];
  const float* b2b = (const float*)d_in[12];
  float* out = (float*)d_out;

  const size_t TB = (size_t)T_DIM * B_DIM;

  uintptr_t p = (uintptr_t)d_ws;
  auto carve = [&](size_t elems) -> unsigned short* {
    p = (p + 255) & ~(uintptr_t)255;
    unsigned short* r = (unsigned short*)p;
    p += elems * sizeof(unsigned short);
    return r;
  };
  unsigned short* Xbf   = carve(TB * I_DIM);
  unsigned short* W1abf = carve((size_t)F1_DIM * I_DIM);
  unsigned short* W1bbf = carve((size_t)H_DIM * F1_DIM);
  unsigned short* Wihbf = carve((size_t)G3H * H_DIM);
  unsigned short* Whhbf = carve((size_t)G3H * H_DIM);
  unsigned short* W2abf = carve((size_t)F2_DIM * H_DIM);
  unsigned short* W2bbf = carve((size_t)O_DIM * F2_DIM);
  unsigned short* A1    = carve(TB * F1_DIM);            // fc1a out, reused for fc2a out
  unsigned short* X2    = carve(TB * H_DIM);             // fc1 output x
  unsigned short* GI    = carve(TB * G3H);               // precomputed x_t @ Wih^T
  unsigned short* HS    = carve((size_t)(T_DIM + 1) * B_DIM * H_DIM);  // h0..hT

  auto conv = [&](const float* s, unsigned short* d, size_t n) {
    int blocks = (int)((n + 255) / 256);
    if (blocks > 8192) blocks = 8192;
    f32_to_bf16_kernel<<<dim3(blocks), dim3(256), 0, stream>>>(s, d, n);
  };
  conv(inputs, Xbf, TB * I_DIM);
  conv(W1a, W1abf, (size_t)F1_DIM * I_DIM);
  conv(W1b, W1bbf, (size_t)H_DIM * F1_DIM);
  conv(Wih, Wihbf, (size_t)G3H * H_DIM);
  conv(Whh, Whhbf, (size_t)G3H * H_DIM);
  conv(W2a, W2abf, (size_t)F2_DIM * H_DIM);
  conv(W2b, W2bbf, (size_t)O_DIM * F2_DIM);

  zero_bf16_kernel<<<dim3(512), dim3(256), 0, stream>>>(HS, (size_t)B_DIM * H_DIM);

  // fc1a: [TB,256] x [1024,256]^T -> relu -> A1 (bf16)
  gemm_bf16_kernel<128,128><<<dim3((int)(TB / 128), F1_DIM / 128), 256, 0, stream>>>(
      Xbf, I_DIM, W1abf, I_DIM, b1a, (int)TB, F1_DIM, I_DIM, A1, nullptr, 1);
  // fc1b: [TB,1024] x [512,1024]^T -> relu -> X2 (bf16)
  gemm_bf16_kernel<128,128><<<dim3((int)(TB / 128), H_DIM / 128), 256, 0, stream>>>(
      A1, F1_DIM, W1bbf, F1_DIM, b1b, (int)TB, H_DIM, F1_DIM, X2, nullptr, 1);
  // gi precompute (hoisted out of the scan): [TB,512] x [1536,512]^T -> GI (bf16)
  gemm_bf16_kernel<128,128><<<dim3((int)(TB / 128), G3H / 128), 256, 0, stream>>>(
      X2, H_DIM, Wihbf, H_DIM, nullptr, (int)TB, G3H, H_DIM, GI, nullptr, 0);

  // GRU scan: 256 sequential step kernels (captured into the graph)
  for (int t = 0; t < T_DIM; ++t) {
    gru_step_kernel<<<dim3(B_DIM / 16, H_DIM / 128), 256, 0, stream>>>(
        GI + (size_t)t * B_DIM * G3H,
        HS + (size_t)t * B_DIM * H_DIM,
        Whhbf, bih, bhh,
        HS + (size_t)(t + 1) * B_DIM * H_DIM);
  }

  // fc2a: [TB,512] x [1024,512]^T -> relu -> A1 (reused, bf16)
  gemm_bf16_kernel<128,128><<<dim3((int)(TB / 128), F2_DIM / 128), 256, 0, stream>>>(
      HS + (size_t)B_DIM * H_DIM, H_DIM, W2abf, H_DIM, b2a, (int)TB, F2_DIM, H_DIM, A1, nullptr, 1);
  // fc2b: [TB,1024] x [64,1024]^T -> +bias -> d_out (f32)
  gemm_bf16_kernel<128,64><<<dim3((int)(TB / 128), O_DIM / 64), 256, 0, stream>>>(
      A1, F2_DIM, W2bbf, F2_DIM, b2b, (int)TB, O_DIM, F2_DIM, nullptr, out, 0);
}
